// WindowAttention3D_3607772529271
// MI455X (gfx1250) — matrix operand order repaired and checked by reading, not compiled, not run on hardware
//
#include <hip/hip_runtime.h>
#include <hip/hip_bf16.h>

typedef __attribute__((ext_vector_type(16))) __bf16 v16bf;
typedef __attribute__((ext_vector_type(8)))  __bf16 v8bf;
typedef __attribute__((ext_vector_type(8)))  float  v8f;

#define NTOK   392
#define KPAD   416          // 13*32, zero-padded token count
#define HD     32
#define HEADS  8
#define BWIN   256
#define DIM    256
#define NCHUNK 13
#define BMSLICE (NTOK*NTOK)   // 153664
#define QSCALE 0.17677669529663687f   // 32^-0.5

static __device__ __forceinline__ v8f wmma_bf16(v16bf a, v16bf b, v8f c) {
    return __builtin_amdgcn_wmma_f32_16x16x32_bf16(false, a, false, b, (short)0, c, false, false);
}

// Build a 16-element fragment from two 16B-aligned 8-half segments.
static __device__ __forceinline__ v16bf ldfrag2(const __bf16* p0, const __bf16* p1) {
    v8bf lo = *(const v8bf*)p0;
    v8bf hi = *(const v8bf*)p1;
    return __builtin_shufflevector(lo, hi, 0,1,2,3,4,5,6,7,8,9,10,11,12,13,14,15);
}

// CDNA5 async global->LDS copy (16B), tracked by ASYNCcnt.
static __device__ __forceinline__ void async_copy_b128(const void* gptr, void* lptr) {
    uint32_t l = (uint32_t)(uintptr_t)lptr;   // low 32 bits of shared aperture = LDS offset
    asm volatile("global_load_async_to_lds_b128 %0, %1, off" :: "v"(l), "v"(gptr) : "memory");
}
static __device__ __forceinline__ void wait_async_le1() {
    asm volatile("s_wait_asynccnt 0x1" ::: "memory");   // oldest copy done (in-order)
}
static __device__ __forceinline__ void wait_async_0() {
    asm volatile("s_wait_asynccnt 0x0" ::: "memory");
}

// ---- prep: x (fp32) -> bf16, vectorized --------------------------------------
__global__ void prep_x(const float* __restrict__ x, __bf16* __restrict__ xb) {
    size_t idx = (size_t)(blockIdx.x * 256 + threadIdx.x) * 8;   // grid covers exactly
    const float4* p = (const float4*)(x + idx);
    float4 a = p[0], b = p[1];
    v8bf t;
    t[0] = (__bf16)a.x; t[1] = (__bf16)a.y; t[2] = (__bf16)a.z; t[3] = (__bf16)a.w;
    t[4] = (__bf16)b.x; t[5] = (__bf16)b.y; t[6] = (__bf16)b.z; t[7] = (__bf16)b.w;
    *(v8bf*)(xb + idx) = t;
}

// ---- prep: weights -> bf16 in B-fragment order -------------------------------
// Bsw[((n16*(K/32) + kblk)*32 + lane)*16 + e] = W[n16*16 + (lane&15)][kblk*32 + 16*(lane>>4) + e]
__global__ void prep_weights(const float* __restrict__ qkv_w, const float* __restrict__ proj_w,
                             __bf16* __restrict__ qkvsw, __bf16* __restrict__ projsw) {
    int idx = blockIdx.x * 256 + threadIdx.x;
    if (idx < 768 * 256) {
        int e = idx & 15, lane = (idx >> 4) & 31, rest = idx >> 9;
        int kblk = rest & 7, n16 = rest >> 3;
        int j = n16 * 16 + (lane & 15);
        int k = kblk * 32 + 16 * (lane >> 4) + e;
        qkvsw[idx] = (__bf16)qkv_w[j * 256 + k];
    } else {
        int i2 = idx - 768 * 256;   // < 256*256 by grid sizing
        int e = i2 & 15, lane = (i2 >> 4) & 31, rest = i2 >> 9;
        int kblk = rest & 7, n16 = rest >> 3;
        int j = n16 * 16 + (lane & 15);
        int k = kblk * 32 + 16 * (lane >> 4) + e;
        projsw[i2] = (__bf16)proj_w[j * 256 + k];
    }
}

// ---- prep: bm[h][w][i][j] = rpb_table[rpi[i][j]][h] + mask[w][i][j] ----------
__global__ void prep_bm(const float* __restrict__ table, const float* __restrict__ mask,
                        const int* __restrict__ rpi, float* __restrict__ bm) {
    int idx = blockIdx.x * 256 + threadIdx.x;      // < 64 * 153664 exactly
    int slice = idx / BMSLICE;
    int rem   = idx - slice * BMSLICE;
    int h = slice >> 3, w = slice & 7;
    bm[idx] = table[rpi[rem] * HEADS + h] + mask[(size_t)w * BMSLICE + rem];
}

// ---- prep: zero the 392..415 token pads of K and V^T -------------------------
__global__ void prep_pads(__bf16* __restrict__ k, __bf16* __restrict__ vT) {
    int bh = blockIdx.x;
    __bf16 z = (__bf16)0.0f;
    for (int idx = threadIdx.x; idx < 768; idx += 256) {   // 24*32
        int tok = 392 + (idx >> 5), d = idx & 31;
        k[(size_t)bh * KPAD * HD + tok * HD + d] = z;
        int dd = idx / 24, col = 392 + (idx - dd * 24);
        vT[(size_t)bh * HD * KPAD + dd * KPAD + col] = z;
    }
}

// ---- QKV GEMM: 64x128 tile, double-buffered async A staging ------------------
__global__ __launch_bounds__(256)
void qkv_gemm(const __bf16* __restrict__ xb, const __bf16* __restrict__ Bsw,
              const float* __restrict__ bias,
              __bf16* __restrict__ q, __bf16* __restrict__ k, __bf16* __restrict__ vT) {
    __shared__ __bf16 sA[2][64 * 32];      // ping-pong A tiles, 64B rows
    const int tid = threadIdx.x;
    const int m0 = blockIdx.y * 64;
    const int n0 = blockIdx.x * 128;
    const int wave = tid >> 5, lane = tid & 31;
    const int mw = (wave & 3) * 16, nw4 = (wave >> 2) * 64;
    const int nn = lane & 15, gg = lane >> 4;
    const int ntile0 = (n0 + nw4) >> 4;
    v8f acc[4];
    #pragma unroll
    for (int t = 0; t < 4; ++t) acc[t] = (v8f){0,0,0,0,0,0,0,0};

    const int arow = tid >> 2, aseg = (tid & 3) * 8;   // 64 rows x 4 chunks of 8 halves
    const __bf16* asrc = xb + (size_t)(m0 + arow) * 256 + aseg;
    const int     aoff = (arow * 32 + aseg) * 2;

    async_copy_b128(asrc, (char*)sA[0] + aoff);        // prologue: tile 0 in flight

    for (int step = 0; step < 8; ++step) {
        const int k0 = step * 32;
        const int cur = step & 1;
        if (step < 7) {
            // issue next tile into the idle buffer (its readers finished at the
            // previous end-of-step barrier), keep it in flight under the WMMAs
            async_copy_b128(asrc + k0 + 32, (char*)sA[1 - cur] + aoff);
            wait_async_le1();                          // current tile complete
        } else {
            wait_async_0();
        }
        __syncthreads();

        const __bf16* arp = &sA[cur][(mw + nn) * 32 + 16 * gg];
        v16bf a = ldfrag2(arp, arp + 8);
        int bi = ((ntile0 * 8 + step) * 32 + lane) * 16;   // fragment-order weights
        #pragma unroll
        for (int t = 0; t < 4; ++t) {
            v16bf bfr = ldfrag2(Bsw + bi + t * 4096, Bsw + bi + t * 4096 + 8);
            acc[t] = wmma_bf16(a, bfr, acc[t]);
        }
        __syncthreads();                               // all reads of sA[cur] done
    }

    #pragma unroll
    for (int r = 0; r < 8; ++r) {
        int i = m0 + mw + gg * 8 + r;
        int b_ = i / NTOK, tok = i - b_ * NTOK;
        #pragma unroll
        for (int half = 0; half < 4; ++half) {
            int j = n0 + nw4 + half * 16 + nn;
            float val = acc[half][r] + bias[j];
            int which = j >> 8, jj = j & 255, hh = jj >> 5, d = jj & 31;
            size_t bh = (size_t)b_ * HEADS + hh;
            if (which == 0)      q[bh * NTOK * HD + tok * HD + d] = (__bf16)(val * QSCALE);
            else if (which == 1) k[bh * KPAD * HD + tok * HD + d] = (__bf16)val;
            else                 vT[bh * HD * KPAD + d * KPAD + tok] = (__bf16)val;
        }
    }
}

// ---- flash attention: one block per (b,h), 8 independent waves ---------------
__global__ __launch_bounds__(256)
void attn_kernel(const __bf16* __restrict__ qg, const __bf16* __restrict__ kg,
                 const __bf16* __restrict__ vg, const float* __restrict__ bm,
                 __bf16* __restrict__ outg) {
    __shared__ __bf16 sK[KPAD * HD];        // K rows (tok, d), pre-padded
    __shared__ __bf16 sVT[HD * KPAD];       // V^T rows (d, tok), pre-padded
    __shared__ __bf16 sP[8][16 * 40];       // per-wave P staging (16x32, stride 40)

    const int bh = blockIdx.x;              // b*8 + h
    const int b = bh >> 3, h = bh & 7, w = b & 7;
    const int tid = threadIdx.x;
    const __bf16* qbase = qg + (size_t)bh * NTOK * HD;
    const float*  bmp   = bm + (size_t)(h * 8 + w) * BMSLICE;

    // async DMA K and V^T into LDS (both are straight 16B-chunk copies)
    {
        const uint4* kg4 = (const uint4*)(kg + (size_t)bh * KPAD * HD);   // 1664 chunks
        const uint4* vg4 = (const uint4*)(vg + (size_t)bh * HD * KPAD);   // 1664 chunks
        for (int idx = tid; idx < 3328; idx += 256) {
            if (idx < 1664) async_copy_b128(kg4 + idx, (char*)sK + idx * 16);
            else            async_copy_b128(vg4 + (idx - 1664), (char*)sVT + (size_t)(idx - 1664) * 16);
        }
        wait_async_0();
    }
    __syncthreads();

    const int wave = tid >> 5, lane = tid & 31;
    const int nn = lane & 15, gg = lane >> 4;
    __bf16* myP = sP[wave];

    for (int qb = wave; qb < 25; qb += 8) {       // ceil(392/16) query blocks
        const int qrow0 = qb * 16;
        // Q A-fragment straight from global (two contiguous 16B segments per lane).
        // Rows >= 392 may read neighbor data; their scores are masked below.
        const __bf16* qp = qbase + (size_t)(qrow0 + nn) * HD;
        v16bf aq = ldfrag2(qp + 16 * gg, qp + 16 * gg + 8);

        v8f o0 = {0,0,0,0,0,0,0,0}, o1 = {0,0,0,0,0,0,0,0};
        float mrow[8], lrow[8];
        #pragma unroll
        for (int r = 0; r < 8; ++r) { mrow[r] = -3.0e38f; lrow[r] = 0.0f; }

        for (int c = 0; c < NCHUNK; ++c) {
            const int t0 = c * 32;
            // score B-frags: contiguous rows of K
            const __bf16* kp0 = &sK[(t0 + nn) * HD + 16 * gg];
            const __bf16* kp1 = &sK[(t0 + 16 + nn) * HD + 16 * gg];
            v16bf bk0 = ldfrag2(kp0, kp0 + 8);
            v16bf bk1 = ldfrag2(kp1, kp1 + 8);
            v8f z = {0,0,0,0,0,0,0,0};
            v8f s0 = wmma_bf16(aq, bk0, z);
            v8f s1 = wmma_bf16(aq, bk1, z);

            const int col0 = t0 + nn, col1 = t0 + 16 + nn;
            #pragma unroll
            for (int r = 0; r < 8; ++r) {
                const int row = qrow0 + gg * 8 + r;
                float x0, x1;
                if (row < NTOK) {
                    x0 = (col0 < NTOK) ? s0[r] + bmp[row * NTOK + col0] : -3.0e38f;
                    x1 = (col1 < NTOK) ? s1[r] + bmp[row * NTOK + col1] : -3.0e38f;
                } else {
                    x0 = (col0 < NTOK) ? 0.0f : -3.0e38f;   // dead rows, discarded
                    x1 = (col1 < NTOK) ? 0.0f : -3.0e38f;
                }
                float mx = fmaxf(x0, x1);
                #pragma unroll
                for (int off = 1; off < 16; off <<= 1)
                    mx = fmaxf(mx, __shfl_xor(mx, off, 32));
                const float mnew = fmaxf(mrow[r], mx);
                const float corr = __expf(mrow[r] - mnew);
                const float p0 = __expf(x0 - mnew);
                const float p1 = __expf(x1 - mnew);
                float ps = p0 + p1;
                #pragma unroll
                for (int off = 1; off < 16; off <<= 1)
                    ps += __shfl_xor(ps, off, 32);
                lrow[r] = lrow[r] * corr + ps;
                o0[r] *= corr;
                o1[r] *= corr;
                mrow[r] = mnew;
                myP[(gg * 8 + r) * 40 + nn]      = (__bf16)p0;
                myP[(gg * 8 + r) * 40 + 16 + nn] = (__bf16)p1;
            }
            // wave-private staging: DS ops are in-order per wave; wait then reload
            asm volatile("s_wait_dscnt 0x0" ::: "memory");

            const __bf16* pp = &myP[nn * 40 + 16 * gg];
            v16bf ap = ldfrag2(pp, pp + 8);
            const __bf16* vp0 = &sVT[nn * KPAD + t0 + 16 * gg];
            const __bf16* vp1 = &sVT[(16 + nn) * KPAD + t0 + 16 * gg];
            v16bf bv0 = ldfrag2(vp0, vp0 + 8);
            v16bf bv1 = ldfrag2(vp1, vp1 + 8);
            o0 = wmma_bf16(ap, bv0, o0);
            o1 = wmma_bf16(ap, bv1, o1);
        }

        #pragma unroll
        for (int r = 0; r < 8; ++r) {
            const int row = qrow0 + gg * 8 + r;
            if (row < NTOK) {
                const float inv = 1.0f / lrow[r];
                size_t o = ((size_t)b * NTOK + row) * DIM + h * HD;
                outg[o + nn]      = (__bf16)(o0[r] * inv);
                outg[o + 16 + nn] = (__bf16)(o1[r] * inv);
            }
        }
    }
}

// ---- proj GEMM: 64x128 tile, double-buffered async A staging -----------------
__global__ __launch_bounds__(256)
void proj_gemm(const __bf16* __restrict__ A, const __bf16* __restrict__ Bsw,
               const float* __restrict__ bias, float* __restrict__ out) {
    __shared__ __bf16 sA[2][64 * 32];
    const int tid = threadIdx.x;
    const int m0 = blockIdx.y * 64;
    const int n0 = blockIdx.x * 128;
    const int wave = tid >> 5, lane = tid & 31;
    const int mw = (wave & 3) * 16, nw4 = (wave >> 2) * 64;
    const int nn = lane & 15, gg = lane >> 4;
    const int ntile0 = (n0 + nw4) >> 4;
    v8f acc[4];
    #pragma unroll
    for (int t = 0; t < 4; ++t) acc[t] = (v8f){0,0,0,0,0,0,0,0};

    const int arow = tid >> 2, aseg = (tid & 3) * 8;
    const __bf16* asrc = A + (size_t)(m0 + arow) * 256 + aseg;
    const int     aoff = (arow * 32 + aseg) * 2;

    async_copy_b128(asrc, (char*)sA[0] + aoff);

    for (int step = 0; step < 8; ++step) {
        const int k0 = step * 32;
        const int cur = step & 1;
        if (step < 7) {
            async_copy_b128(asrc + k0 + 32, (char*)sA[1 - cur] + aoff);
            wait_async_le1();
        } else {
            wait_async_0();
        }
        __syncthreads();

        const __bf16* arp = &sA[cur][(mw + nn) * 32 + 16 * gg];
        v16bf a = ldfrag2(arp, arp + 8);
        int bi = ((ntile0 * 8 + step) * 32 + lane) * 16;
        #pragma unroll
        for (int t = 0; t < 4; ++t) {
            v16bf bfr = ldfrag2(Bsw + bi + t * 4096, Bsw + bi + t * 4096 + 8);
            acc[t] = wmma_bf16(a, bfr, acc[t]);
        }
        __syncthreads();
    }

    #pragma unroll
    for (int r = 0; r < 8; ++r) {
        size_t i = (size_t)(m0 + mw + gg * 8 + r);
        #pragma unroll
        for (int half = 0; half < 4; ++half) {
            int j = n0 + nw4 + half * 16 + nn;
            out[i * 256 + j] = acc[half][r] + bias[j];
        }
    }
}

// ---- host --------------------------------------------------------------------
extern "C" void kernel_launch(void* const* d_in, const int* in_sizes, int n_in,
                              void* d_out, int out_size, void* d_ws, size_t ws_size,
                              hipStream_t stream) {
    (void)in_sizes; (void)n_in; (void)out_size; (void)ws_size;
    const float* x      = (const float*)d_in[0];
    const float* mask   = (const float*)d_in[1];
    const float* table  = (const float*)d_in[2];
    const float* qkv_w  = (const float*)d_in[3];
    const float* qkv_b  = (const float*)d_in[4];
    const float* proj_w = (const float*)d_in[5];
    const float* proj_b = (const float*)d_in[6];
    const int*   rpi    = (const int*)d_in[7];
    float* out = (float*)d_out;

    const size_t NBH = (size_t)BWIN * HEADS;                 // 2048
    char* ws = (char*)d_ws;
    __bf16* xb      = (__bf16*)ws;  ws += (size_t)BWIN * NTOK * DIM * 2;   // 51.4 MB
    __bf16* q_ws    = (__bf16*)ws;  ws += NBH * NTOK * HD * 2;             // 51.4 MB
    __bf16* k_ws    = (__bf16*)ws;  ws += NBH * KPAD * HD * 2;             // 54.5 MB
    __bf16* vT_ws   = (__bf16*)ws;  ws += NBH * HD * KPAD * 2;             // 54.5 MB
    __bf16* attnout = (__bf16*)ws;  ws += (size_t)BWIN * NTOK * DIM * 2;   // 51.4 MB
    __bf16* qkvsw   = (__bf16*)ws;  ws += (size_t)768 * 256 * 2;
    __bf16* projsw  = (__bf16*)ws;  ws += (size_t)256 * 256 * 2;
    float*  bm      = (float*)ws;   // 64 * 153664 * 4 = 39.3 MB

    prep_x<<<12544, 256, 0, stream>>>(x, xb);                 // 25,690,112 / 8 / 256
    prep_weights<<<1024, 256, 0, stream>>>(qkv_w, proj_w, qkvsw, projsw);
    prep_bm<<<(64 * BMSLICE) / 256, 256, 0, stream>>>(table, mask, rpi, bm);
    prep_pads<<<NBH, 256, 0, stream>>>(k_ws, vT_ws);
    qkv_gemm<<<dim3(6, 1568), 256, 0, stream>>>(xb, qkvsw, qkv_b, q_ws, k_ws, vT_ws);
    attn_kernel<<<NBH, 256, 0, stream>>>(q_ws, k_ws, vT_ws, bm, attnout);
    proj_gemm<<<dim3(2, 1568), 256, 0, stream>>>(attnout, projsw, proj_b, out);
}
